// GroupedQueryAttention_82764019794558
// MI455X (gfx1250) — matrix operand order repaired and checked by reading, not compiled, not run on hardware
//
#include <hip/hip_runtime.h>
#include <hip/hip_bf16.h>

// ---------------------------------------------------------------------------
// GQA block for MI455X (gfx1250): bf16 WMMA 16x16x32, f32 accumulate,
// async global->LDS double-buffered staging (ASYNCcnt), DPP16 softmax
// reductions, bf16 workspace.
// D_MODEL=1024, HEADS=16, KV_HEADS=4, D_K=64, B=2, S=2048  (M = B*S = 4096)
// ---------------------------------------------------------------------------

typedef __attribute__((ext_vector_type(16))) __bf16 v16bf;
typedef __attribute__((ext_vector_type(8)))  float  v8f;
typedef __attribute__((ext_vector_type(4)))  int    v4i;

#define WMMA_BF16(A, Bm, C) \
  __builtin_amdgcn_wmma_f32_16x16x32_bf16(false, (A), false, (Bm), (short)0, (C), false, false)

#define SEQ 2048

// ---- gfx1250 async global->LDS copy (16B per lane), ASYNCcnt tracked ------
__device__ __forceinline__ void async_load_b128(const __bf16* g, __bf16* l) {
#if defined(__HIP_DEVICE_COMPILE__) && __has_builtin(__builtin_amdgcn_global_load_async_to_lds_b128)
  __builtin_amdgcn_global_load_async_to_lds_b128(
      (__attribute__((address_space(1))) v4i*)g,
      (__attribute__((address_space(3))) v4i*)l, 0, 0);
#else
  unsigned lds_off = (unsigned)(unsigned long long)
      (__attribute__((address_space(3))) char*)l;
  asm volatile("global_load_async_to_lds_b128 %0, %1, off"
               :: "v"(lds_off), "v"(g) : "memory");
#endif
}

__device__ __forceinline__ void wait_asynccnt0() {
#if defined(__HIP_DEVICE_COMPILE__) && __has_builtin(__builtin_amdgcn_s_wait_asynccnt)
  __builtin_amdgcn_s_wait_asynccnt(0);
#else
  asm volatile("s_wait_asynccnt 0x0" ::: "memory");
#endif
}

// ---- DPP16 butterfly reduction across each 16-lane half-wave (pure VALU) --
template <int CTRL>
__device__ __forceinline__ float dpp_xchg(float x) {
  return __int_as_float(__builtin_amdgcn_update_dpp(
      0, __float_as_int(x), CTRL, 0xF, 0xF, true));
}
__device__ __forceinline__ float rowmax16(float v) {
  v = fmaxf(v, dpp_xchg<0xB1>(v));    // quad_perm [1,0,3,2]  (xor 1)
  v = fmaxf(v, dpp_xchg<0x4E>(v));    // quad_perm [2,3,0,1]  (xor 2)
  v = fmaxf(v, dpp_xchg<0x141>(v));   // row_half_mirror      (cross quads)
  v = fmaxf(v, dpp_xchg<0x140>(v));   // row_mirror           (cross 8s)
  return v;
}
__device__ __forceinline__ float rowsum16(float v) {
  v += dpp_xchg<0xB1>(v);
  v += dpp_xchg<0x4E>(v);
  v += dpp_xchg<0x141>(v);
  v += dpp_xchg<0x140>(v);
  return v;
}

// ---------------------------------------------------------------------------
// Pre-pack kernels (run once, bandwidth-trivial: ~26 MB total @ 23.3 TB/s)
// ---------------------------------------------------------------------------
__global__ __launch_bounds__(256) void cvt_f32_bf16_kernel(
    const float* __restrict__ src, __bf16* __restrict__ dst) {
  const int i = (blockIdx.x * 256 + threadIdx.x) * 4;
  const float4 v = *(const float4*)(src + i);
  dst[i + 0] = (__bf16)v.x;
  dst[i + 1] = (__bf16)v.y;
  dst[i + 2] = (__bf16)v.z;
  dst[i + 3] = (__bf16)v.w;
}

// W[K][N] fp32 -> Wt[N][K] bf16  (64x64 LDS tile transpose)
__global__ __launch_bounds__(256) void cvt_transpose_w_kernel(
    const float* __restrict__ W, __bf16* __restrict__ Wt, int K, int N) {
  __shared__ __bf16 t[64][65];
  const int tid = threadIdx.x;
  const int k0 = blockIdx.y * 64, n0 = blockIdx.x * 64;
#pragma unroll
  for (int i = 0; i < 16; ++i) {
    int idx = i * 256 + tid;
    int kl = idx >> 6, nl = idx & 63;
    t[kl][nl] = (__bf16)W[(size_t)(k0 + kl) * N + (n0 + nl)];
  }
  __syncthreads();
#pragma unroll
  for (int i = 0; i < 16; ++i) {
    int idx = i * 256 + tid;
    int nl = idx >> 6, kl = idx & 63;
    Wt[(size_t)(n0 + nl) * K + (k0 + kl)] = t[kl][nl];
  }
}

// Vw[4096][256] bf16 -> Vt[8][64][2048] bf16  (per (b,kv_head): [dk][s])
__global__ __launch_bounds__(256) void transpose_v_kernel(
    const __bf16* __restrict__ Vw, __bf16* __restrict__ Vt) {
  __shared__ __bf16 t[64][65];
  const int tid = threadIdx.x;
  const int bh = blockIdx.y;                 // b*4 + kvh
  const int b = bh >> 2, kvh = bh & 3;
  const int s0 = blockIdx.x * 64;
#pragma unroll
  for (int i = 0; i < 16; ++i) {
    int idx = i * 256 + tid;
    int sl = idx >> 6, dk = idx & 63;
    t[sl][dk] = Vw[(size_t)(b * SEQ + s0 + sl) * 256 + kvh * 64 + dk];
  }
  __syncthreads();
#pragma unroll
  for (int i = 0; i < 16; ++i) {
    int idx = i * 256 + tid;
    int dk = idx >> 6, sl = idx & 63;
    Vt[((size_t)bh * 64 + dk) * SEQ + s0 + sl] = t[sl][dk];
  }
}

// ---------------------------------------------------------------------------
// GEMM + bias:  C[M,N] = A[M,K] @ Wt[N,K]^T + bias[N]
// A, Wt bf16; C float or bf16.  Tile 128(M) x 64(N), BK = 64 (2 WMMA K-steps
// per barrier).  Double-buffered async staging; 8 waves x (16x64) strips.
// ---------------------------------------------------------------------------
#define TM 128
#define TN 64
#define TK 64

template <typename TC>
__global__ __launch_bounds__(256) void gemm_bias_kernel(
    const __bf16* __restrict__ A, const __bf16* __restrict__ Wt,
    const float* __restrict__ bias, TC* __restrict__ C,
    int M, int N, int Kdim)
{
  __shared__ __align__(16) __bf16 Alds[2][TM][TK];   // 2 x 16 KB
  __shared__ __align__(16) __bf16 Blds[2][TN][TK];   // 2 x  8 KB

  const int tid  = threadIdx.x;
  const int wave = tid >> 5;
  const int lane = tid & 31;
  const int hi   = lane >> 4;
  const int ln   = lane & 15;

  const int m0 = blockIdx.y * TM;
  const int n0 = blockIdx.x * TN;
  const int mw = wave * 16;

  // 48 x b128 async ops per K-step (A: 32, B: 16) -> 6 per wave, uniform.
  auto stage = [&](int c, int buf) {
    const int k0 = c * TK;
#pragma unroll
    for (int i = 0; i < 6; ++i) {
      const int slot = wave * 6 + i;          // 0..47
      if (slot < 32) {                        // A tile: 128 rows x 128B
        const int e = slot * 32 + lane;       // 0..1023
        const int r = e >> 3, c16 = e & 7;
        async_load_b128(A + (size_t)(m0 + r) * Kdim + k0 + c16 * 8,
                        &Alds[buf][r][c16 * 8]);
      } else {                                // B tile: 64 rows x 128B
        const int e = (slot - 32) * 32 + lane; // 0..511
        const int r = e >> 3, c16 = e & 7;
        async_load_b128(Wt + (size_t)(n0 + r) * Kdim + k0 + c16 * 8,
                        &Blds[buf][r][c16 * 8]);
      }
    }
  };

  v8f acc[4];
#pragma unroll
  for (int j = 0; j < 4; ++j)
#pragma unroll
    for (int r = 0; r < 8; ++r) acc[j][r] = 0.0f;

  const int NC = Kdim / TK;
  stage(0, 0);
  for (int c = 0; c < NC; ++c) {
    const int buf = c & 1;
    wait_asynccnt0();          // this wave's tile-c loads complete
    __syncthreads();           // all waves' data visible; prev compute done
    if (c + 1 < NC) stage(c + 1, buf ^ 1);   // prefetch overlaps compute

#pragma unroll
    for (int ks = 0; ks < 2; ++ks) {
      // A fragment (ISA 16-bit A layout)
      v16bf af;
      {
        const int m = mw + ln;
#pragma unroll
        for (int v = 0; v < 8; ++v) {
          const int kk = ks * 32 + ((v < 4) ? (2 * v) : (8 + 2 * v)) + hi * 8;
          af[2 * v]     = Alds[buf][m][kk];
          af[2 * v + 1] = Alds[buf][m][kk + 1];
        }
      }
#pragma unroll
      for (int j = 0; j < 4; ++j) {
        v16bf bf;
        const int n = j * 16 + ln;
#pragma unroll
        for (int v = 0; v < 8; ++v) {
          const int kk = ks * 32 + ((v < 4) ? (2 * v) : (8 + 2 * v)) + hi * 8;
          bf[2 * v]     = Blds[buf][n][kk];
          bf[2 * v + 1] = Blds[buf][n][kk + 1];
        }
        acc[j] = WMMA_BF16(af, bf, acc[j]);
      }
    }
  }

  // C layout: VGPR r -> M = r + hi*8, N = ln
#pragma unroll
  for (int j = 0; j < 4; ++j) {
    const int n = n0 + j * 16 + ln;
    const float bv = bias[n];
#pragma unroll
    for (int r = 0; r < 8; ++r) {
      const int m = m0 + mw + r + hi * 8;
      C[(size_t)m * N + n] = (TC)(acc[j][r] + bv);
    }
  }
}

// ---------------------------------------------------------------------------
// Flash attention for GQA.  Workgroup = (b, kv_head, 32-query block).
// Wave w -> head-in-group g = w&3, query sub-tile (w>>2)*16.
// 64-key chunks async double-buffered in LDS, shared by all 4 heads.
// Softmax in log2 domain; DPP16 row reductions (no LDS traffic).
// Q:[4096,1024]  K:[4096,256]  Vt:[8][64][2048]  O:[4096,1024]  (all bf16)
// ---------------------------------------------------------------------------
#define CK 64   // keys per chunk

__global__ __launch_bounds__(256) void gqa_attn_kernel(
    const __bf16* __restrict__ Q, const __bf16* __restrict__ K,
    const __bf16* __restrict__ Vt, __bf16* __restrict__ O)
{
  __shared__ __align__(16) __bf16 Klds[2][CK][64];   // [key][dk]  2 x 8 KB
  __shared__ __align__(16) __bf16 Vlds[2][64][CK];   // [dk][key]  2 x 8 KB
  __shared__ __align__(16) __bf16 Plds[8][16][CK];   // per-wave P  16 KB

  const int tid  = threadIdx.x;
  const int wave = tid >> 5;
  const int lane = tid & 31;
  const int hi   = lane >> 4;
  const int ln   = lane & 15;

  const int qblk = blockIdx.x;   // 0..63
  const int kvh  = blockIdx.y;   // 0..3
  const int b    = blockIdx.z;   // 0..1

  const int g  = wave & 3;
  const int qw = (wave >> 2) * 16;
  const int h  = kvh * 4 + g;               // group-major head id
  const int q0 = qblk * 32 + qw;
  const size_t rowQ = (size_t)(b * SEQ + q0);

  const __bf16* Kg  = K  + (size_t)(b * SEQ) * 256 + kvh * 64;
  const __bf16* Vtg = Vt + (size_t)(b * 4 + kvh) * 64 * SEQ;

  // 32 x b128 async ops per chunk (K: 16, V: 16) -> 4 per wave, uniform.
  auto stage = [&](int c, int buf) {
    const int sk = c * CK;
#pragma unroll
    for (int i = 0; i < 4; ++i) {
      const int slot = wave * 4 + i;           // 0..31
      if (slot < 16) {                         // K chunk: 64 rows x 128B
        const int e = slot * 32 + lane;        // 0..511
        const int r = e >> 3, c16 = e & 7;
        async_load_b128(Kg + (size_t)(sk + r) * 256 + c16 * 8,
                        &Klds[buf][r][c16 * 8]);
      } else {                                 // V chunk: 64 rows x 128B
        const int e = (slot - 16) * 32 + lane; // 0..511
        const int r = e >> 3, c16 = e & 7;
        async_load_b128(Vtg + (size_t)r * SEQ + sk + c16 * 8,
                        &Vlds[buf][r][c16 * 8]);
      }
    }
  };

  // Preload Q fragments (contraction d_k = 64 = 2 x 32)
  v16bf qf[2];
  {
    const __bf16* qp = Q + (rowQ + ln) * 1024 + h * 64;
#pragma unroll
    for (int ks = 0; ks < 2; ++ks)
#pragma unroll
      for (int v = 0; v < 8; ++v) {
        const int kk = ks * 32 + ((v < 4) ? (2 * v) : (8 + 2 * v)) + hi * 8;
        qf[ks][2 * v]     = qp[kk];
        qf[ks][2 * v + 1] = qp[kk + 1];
      }
  }

  v8f oacc[4];
  float mrow[8], lrow[8];
#pragma unroll
  for (int r = 0; r < 8; ++r) { mrow[r] = -3.0e38f; lrow[r] = 0.0f; }
#pragma unroll
  for (int j = 0; j < 4; ++j)
#pragma unroll
    for (int r = 0; r < 8; ++r) oacc[j][r] = 0.0f;

  // scores scaled to log2 domain: 1/sqrt(64) * log2(e)
  const float SCL = 0.125f * 1.44269504088896340736f;

  const int NC = SEQ / CK;      // 32 chunks
  stage(0, 0);
  for (int c = 0; c < NC; ++c) {
    const int buf = c & 1;
    wait_asynccnt0();
    __syncthreads();
    if (c + 1 < NC) stage(c + 1, buf ^ 1);

    // ---- scores: S[16q x 64k] = Q(16x64) @ K^T -> four 16x16 C tiles ----
    v8f st[4];
#pragma unroll
    for (int t = 0; t < 4; ++t) {
#pragma unroll
      for (int r = 0; r < 8; ++r) st[t][r] = 0.0f;
#pragma unroll
      for (int ks = 0; ks < 2; ++ks) {
        v16bf bf;
        const int n = t * 16 + ln;                 // key within chunk
#pragma unroll
        for (int v = 0; v < 8; ++v) {
          const int kk = ks * 32 + ((v < 4) ? (2 * v) : (8 + 2 * v)) + hi * 8;  // dk
          bf[2 * v]     = Klds[buf][n][kk];
          bf[2 * v + 1] = Klds[buf][n][kk + 1];
        }
        st[t] = WMMA_BF16(qf[ks], bf, st[t]);
      }
    }

    // ---- online softmax (log2 domain; DPP16 reductions, pure VALU) ----
#pragma unroll
    for (int r = 0; r < 8; ++r) {
      float s[4];
      float cm = -3.0e38f;
#pragma unroll
      for (int t = 0; t < 4; ++t) {
        s[t] = st[t][r] * SCL;
        cm = fmaxf(cm, s[t]);
      }
      cm = rowmax16(cm);
      const float mn    = fmaxf(mrow[r], cm);
      const float alpha = exp2f(mrow[r] - mn);
      float rs = 0.0f;
#pragma unroll
      for (int t = 0; t < 4; ++t) {
        s[t] = exp2f(s[t] - mn);
        rs += s[t];
      }
      rs = rowsum16(rs);
      lrow[r] = lrow[r] * alpha + rs;
      mrow[r] = mn;
#pragma unroll
      for (int j = 0; j < 4; ++j) oacc[j][r] *= alpha;
#pragma unroll
      for (int t = 0; t < 4; ++t) st[t][r] = s[t];
    }

    // ---- C-layout -> A-layout for P via per-wave LDS scratch ----
#pragma unroll
    for (int t = 0; t < 4; ++t)
#pragma unroll
      for (int r = 0; r < 8; ++r)
        Plds[wave][r + hi * 8][t * 16 + ln] = (__bf16)st[t][r];

    v16bf pf[2];
#pragma unroll
    for (int ks = 0; ks < 2; ++ks)
#pragma unroll
      for (int v = 0; v < 8; ++v) {
        const int kk = ks * 32 + ((v < 4) ? (2 * v) : (8 + 2 * v)) + hi * 8;
        pf[ks][2 * v]     = Plds[wave][ln][kk];
        pf[ks][2 * v + 1] = Plds[wave][ln][kk + 1];
      }

    // ---- O += P(16x64) @ V(64x64) : 8 WMMAs ----
#pragma unroll
    for (int j = 0; j < 4; ++j) {
      const int n = j * 16 + ln;                   // dk
#pragma unroll
      for (int ks = 0; ks < 2; ++ks) {
        v16bf vf;
#pragma unroll
        for (int v = 0; v < 8; ++v) {
          const int kk = ks * 32 + ((v < 4) ? (2 * v) : (8 + 2 * v)) + hi * 8;  // key
          vf[2 * v]     = Vlds[buf][n][kk];
          vf[2 * v + 1] = Vlds[buf][n][kk + 1];
        }
        oacc[j] = WMMA_BF16(pf[ks], vf, oacc[j]);
      }
    }
  }

  // ---- normalize + store bf16 (b, s, h, dk) == row-major [4096,1024] ----
#pragma unroll
  for (int j = 0; j < 4; ++j) {
    const int n = j * 16 + ln;
#pragma unroll
    for (int r = 0; r < 8; ++r) {
      const int m = r + hi * 8;
      O[(rowQ + m) * 1024 + h * 64 + n] = (__bf16)(oacc[j][r] / lrow[r]);
    }
  }
}

// ---------------------------------------------------------------------------
// Launch: pre-pack -> Q/K/V projections -> V transpose -> attention -> out.
// Workspace (bf16 elements):
//   xb 4M | Wqt 1M | Wkt 256K | Wvt 256K | Wot 1M | Qw 4M | Kw 1M | Vw 1M
//   | Vt 1M | Ow 4M   (~35 MB total)
// ---------------------------------------------------------------------------
extern "C" void kernel_launch(void* const* d_in, const int* in_sizes, int n_in,
                              void* d_out, int out_size, void* d_ws, size_t ws_size,
                              hipStream_t stream) {
  const float* x  = (const float*)d_in[0];
  const float* Wq = (const float*)d_in[1];
  const float* bq = (const float*)d_in[2];
  const float* Wk = (const float*)d_in[3];
  const float* bk = (const float*)d_in[4];
  const float* Wv = (const float*)d_in[5];
  const float* bv = (const float*)d_in[6];
  const float* Wo = (const float*)d_in[7];
  const float* bo = (const float*)d_in[8];
  float* out = (float*)d_out;

  __bf16* xb  = (__bf16*)d_ws;
  __bf16* Wqt = xb  + (size_t)4096 * 1024;
  __bf16* Wkt = Wqt + (size_t)1024 * 1024;
  __bf16* Wvt = Wkt + (size_t)256 * 1024;
  __bf16* Wot = Wvt + (size_t)256 * 1024;
  __bf16* Qw  = Wot + (size_t)1024 * 1024;
  __bf16* Kw  = Qw  + (size_t)4096 * 1024;
  __bf16* Vw  = Kw  + (size_t)4096 * 256;
  __bf16* Vt  = Vw  + (size_t)4096 * 256;
  __bf16* Ow  = Vt  + (size_t)4096 * 256;

  const dim3 blk(256);
  // pre-pack
  cvt_f32_bf16_kernel<<<4096, blk, 0, stream>>>(x, xb);
  cvt_transpose_w_kernel<<<dim3(16, 16), blk, 0, stream>>>(Wq, Wqt, 1024, 1024);
  cvt_transpose_w_kernel<<<dim3(4, 16),  blk, 0, stream>>>(Wk, Wkt, 1024, 256);
  cvt_transpose_w_kernel<<<dim3(4, 16),  blk, 0, stream>>>(Wv, Wvt, 1024, 256);
  cvt_transpose_w_kernel<<<dim3(16, 16), blk, 0, stream>>>(Wo, Wot, 1024, 1024);
  // projections (bf16 out)
  gemm_bias_kernel<__bf16><<<dim3(1024 / TN, 4096 / TM), blk, 0, stream>>>(xb, Wqt, bq, Qw, 4096, 1024, 1024);
  gemm_bias_kernel<__bf16><<<dim3(256 / TN,  4096 / TM), blk, 0, stream>>>(xb, Wkt, bk, Kw, 4096, 256, 1024);
  gemm_bias_kernel<__bf16><<<dim3(256 / TN,  4096 / TM), blk, 0, stream>>>(xb, Wvt, bv, Vw, 4096, 256, 1024);
  // V -> per-head [dk][s]
  transpose_v_kernel<<<dim3(32, 8), blk, 0, stream>>>(Vw, Vt);
  // attention
  gqa_attn_kernel<<<dim3(64, 4, 2), blk, 0, stream>>>(Qw, Kw, Vt, Ow);
  // out = O @ Wo + bo (fp32 out)
  gemm_bias_kernel<float><<<dim3(1024 / TN, 4096 / TM), blk, 0, stream>>>(Ow, Wot, bo, out, 4096, 1024, 1024);
}
